// PNAnet_80264348827991
// MI455X (gfx1250) — compile-verified, matrix operand order checked by reading, not run on hardware
//
#include <hip/hip_runtime.h>
#include <hip/hip_bf16.h>
#include <math.h>

#define N_NODES 32768
#define N_EDGES 524288
#define N_GRAPHS 64
#define HID 128

typedef __attribute__((ext_vector_type(16))) __bf16 v16bf;
typedef __attribute__((ext_vector_type(8)))  __bf16 v8bf;
typedef __attribute__((ext_vector_type(8)))  float  v8f;
typedef __attribute__((__vector_size__(4 * sizeof(int)))) int v4i;

#if defined(__has_builtin)
# if __has_builtin(__builtin_amdgcn_global_load_async_to_lds_b128)
#  define HAS_ASYNC_LDS 1
# endif
#endif
#ifndef HAS_ASYNC_LDS
# define HAS_ASYNC_LDS 0
#endif

#define MODE_EDGE  0
#define MODE_FEAT  1
#define MODE_PLAIN 2

// ---- ordered-uint encoding so float min/max can use integer atomics ----
__device__ __forceinline__ unsigned fenc(float f) {
    unsigned u = __float_as_uint(f);
    return (u & 0x80000000u) ? ~u : (u | 0x80000000u);
}
__device__ __forceinline__ float fdec(unsigned e) {
    unsigned u = (e & 0x80000000u) ? (e & 0x7FFFFFFFu) : ~e;
    return __uint_as_float(u);
}
__device__ __forceinline__ float leaky(float v) { return v > 0.f ? v : 0.01f * v; }

__device__ __forceinline__ void async_wait0() {
#if defined(__has_builtin)
# if __has_builtin(__builtin_amdgcn_s_wait_asynccnt)
    __builtin_amdgcn_s_wait_asynccnt(0);
    return;
# endif
#endif
    asm volatile("s_wait_asynccnt 0x0" ::: "memory");
}

// ------------------------------ utility kernels ------------------------------
__global__ void k_fill_f(float* p, float v, int n) {
    int i = blockIdx.x * blockDim.x + threadIdx.x;
    if (i < n) p[i] = v;
}
__global__ void k_fill_u(unsigned* p, unsigned v, int n) {
    int i = blockIdx.x * blockDim.x + threadIdx.x;
    if (i < n) p[i] = v;
}

__global__ void k_degree(const int* __restrict__ dst, float* __restrict__ cnt, int E) {
    int i = blockIdx.x * blockDim.x + threadIdx.x;
    if (i < E) atomicAdd(&cnt[dst[i]], 1.0f);
}

__global__ void k_redstats(const float* __restrict__ cnt, float* __restrict__ red, int N) {
    __shared__ float s0[256], s1[256];
    float a = 0.f, b = 0.f;
    for (int i = blockIdx.x * blockDim.x + threadIdx.x; i < N; i += gridDim.x * blockDim.x) {
        float c = cnt[i];
        a += logf(c + 1.0f);
        b += c;
    }
    s0[threadIdx.x] = a; s1[threadIdx.x] = b;
    __syncthreads();
    for (int o = 128; o > 0; o >>= 1) {
        if ((int)threadIdx.x < o) { s0[threadIdx.x] += s0[threadIdx.x + o]; s1[threadIdx.x] += s1[threadIdx.x + o]; }
        __syncthreads();
    }
    if (threadIdx.x == 0) { atomicAdd(&red[0], s0[0]); atomicAdd(&red[1], s1[0]); }
}

__global__ void k_scalers(const float* __restrict__ cnt, const float* __restrict__ red,
                          float* __restrict__ amp, float* __restrict__ att,
                          float* __restrict__ lsc, float* __restrict__ invd, int N) {
    int i = blockIdx.x * blockDim.x + threadIdx.x;
    if (i >= N) return;
    float avg_log = red[0] / (float)N;
    float avg_lin = red[1] / (float)N;
    float deg = fmaxf(cnt[i], 1.0f);
    float ld = logf(deg + 1.0f);
    amp[i]  = ld / avg_log;
    att[i]  = avg_log / ld;
    lsc[i]  = deg / avg_lin;
    invd[i] = 1.0f / deg;
}

// W [K][Nout] f32 (row-major din x dout)  ->  Bt [Npad][Kpad] bf16, zero padded
__global__ void k_convW(const float* __restrict__ W, __bf16* __restrict__ Bt,
                        int K, int Nout, int Kpad, int Npad) {
    int i = blockIdx.x * blockDim.x + threadIdx.x;
    int tot = Npad * Kpad;
    if (i >= tot) return;
    int n = i / Kpad, k = i - n * Kpad;
    float v = (n < Nout && k < K) ? W[(size_t)k * Nout + n] : 0.0f;
    Bt[i] = (__bf16)v;
}

// ------------------------------ core WMMA GEMM ------------------------------
// Block = 4 waves; each wave owns a 16-row tile and computes 16 x (NT*16) of
// D = A x B + bias.  Per 32-wide K chunk:
//   * each wave stages its A chunk into LDS (bf16), using uniform per-region
//     vectorized copies when F % 32 == 0 (the generic per-element path is only
//     taken for layer 0, F = 4);
//   * the B chunk (NT*16 x 32 bf16) is staged into LDS ONCE per block --
//     via GLOBAL_LOAD_ASYNC_TO_LDS_B128 when the toolchain exposes it --
//     and shared by all 4 waves (4x less L2 traffic, copy overlapped with
//     the A-staging ALU work, completion via s_wait_asynccnt + barrier);
//   * NT back-to-back v_wmma_f32_16x16x32_bf16 reuse the staged A fragment.
template <int MODE, int NT>
__global__ void __launch_bounds__(128)
k_gemm(const float* __restrict__ A0,        // EDGE/FEAT: h [N][F]; PLAIN: A [rows][Kreal]
       const int* __restrict__ dst, const int* __restrict__ src,
       const float* __restrict__ eattr,     // [E][4]
       const float* __restrict__ eW,        // [4][F]
       const float* __restrict__ eB,        // [F]
       const float* __restrict__ ssum, const float* __restrict__ ssq,
       const unsigned* __restrict__ smn, const unsigned* __restrict__ smx,
       const float* __restrict__ amp, const float* __restrict__ att,
       const float* __restrict__ lsc, const float* __restrict__ invd,
       const float* __restrict__ cnt,
       const __bf16* __restrict__ Bt,       // [Npad][Kpad]
       const float* __restrict__ bias,      // [Nout]
       float* __restrict__ out,             // FEAT/PLAIN store [rows][HID]
       float* __restrict__ osum, float* __restrict__ osq,
       unsigned* __restrict__ omn, unsigned* __restrict__ omx,
       int F, int Kreal, int Kpad, int Nout)
{
    __shared__ __align__(16) __bf16 As[4][16][32];
    __shared__ __align__(16) __bf16 Bs[NT * 16][32];
    const int lane = threadIdx.x & 31;
    const int w    = threadIdx.x >> 5;
    const int wid  = blockIdx.x * 4 + w;      // grids divide exactly: no guard
    const int row0 = wid << 4;
    const int M    = lane & 15;            // fragment row / B column within tile
    const int kb   = (lane >> 4) << 3;     // fragment K base per ISA 16-bit layout

    // staging assignment: lane fills row r, 16 contiguous columns at cbase
    const int r     = lane >> 1;
    const int cbase = (lane & 1) << 4;
    const int rowA  = row0 + r;

    // per-lane row-invariant values
    int   dIdx = 0, sIdx = 0;
    float ea0 = 0.f, ea1 = 0.f, ea2 = 0.f, ea3 = 0.f;
    float idv = 0.f, scl1 = 0.f, scl2 = 0.f, scl3 = 0.f;
    bool  has = false;
    const float* Arow = nullptr;
    if (MODE == MODE_EDGE) {
        dIdx = dst[rowA]; sIdx = src[rowA];
        ea0 = eattr[(size_t)rowA * 4 + 0];
        ea1 = eattr[(size_t)rowA * 4 + 1];
        ea2 = eattr[(size_t)rowA * 4 + 2];
        ea3 = eattr[(size_t)rowA * 4 + 3];
    } else if (MODE == MODE_FEAT) {
        idv = invd[rowA]; has = cnt[rowA] > 0.f;
        scl1 = amp[rowA]; scl2 = att[rowA]; scl3 = lsc[rowA];
    } else {
        Arow = A0 + (size_t)rowA * Kreal;
    }

    const bool fast = (F & 31) == 0;
    v8f acc[NT] = {};

    for (int k0 = 0; k0 < Kpad; k0 += 32) {
        // ---------- B chunk: cooperative block-wide stage into LDS ----------
        for (int cidx = threadIdx.x; cidx < NT * 64; cidx += 128) {
            int n = cidx >> 2, q = cidx & 3;
            const __bf16* g = Bt + (size_t)n * Kpad + k0 + (q << 3);
            __bf16* l = &Bs[n][q << 3];
#if HAS_ASYNC_LDS
            __builtin_amdgcn_global_load_async_to_lds_b128((v4i*)g, (v4i*)l, 0, 0);
#else
            *(v8bf*)l = *(const v8bf*)g;
#endif
        }

        // ---------------- A chunk: per-wave stage into LDS ----------------
        if (MODE == MODE_PLAIN) {
            const float* p = Arow + k0 + cbase;
#pragma unroll
            for (int i = 0; i < 16; ++i) As[w][r][cbase + i] = (__bf16)p[i];
        } else if (fast) {
            if (MODE == MODE_EDGE) {
                if (k0 + 32 <= F) {
                    const float* p = A0 + (size_t)dIdx * F + k0 + cbase;
#pragma unroll
                    for (int i = 0; i < 16; ++i) As[w][r][cbase + i] = (__bf16)p[i];
                } else if (k0 + 32 <= 2 * F) {
                    const float* p = A0 + (size_t)sIdx * F + (k0 - F) + cbase;
#pragma unroll
                    for (int i = 0; i < 16; ++i) As[w][r][cbase + i] = (__bf16)p[i];
                } else {
                    int j0 = (k0 - 2 * F) + cbase;
#pragma unroll
                    for (int i = 0; i < 16; ++i) {
                        int j = j0 + i;
                        float v = eB[j] + ea0 * eW[j] + ea1 * eW[F + j]
                                        + ea2 * eW[2 * F + j] + ea3 * eW[3 * F + j];
                        As[w][r][cbase + i] = (__bf16)v;
                    }
                }
            } else { // MODE_FEAT, F multiple of 32: chunk is one (grp,stat) block
                if (k0 + 32 <= F) {
                    const float* p = A0 + (size_t)rowA * F + k0 + cbase;
#pragma unroll
                    for (int i = 0; i < 16; ++i) As[w][r][cbase + i] = (__bf16)p[i];
                } else {
                    int kp0  = k0 - F;
                    int grp  = kp0 / (4 * F);
                    int rem  = kp0 - grp * (4 * F);
                    int stat = rem / F;
                    int f0   = (rem - stat * F) + cbase;
                    float sc = (grp == 0) ? 1.f : (grp == 1) ? scl1 : (grp == 2) ? scl2 : scl3;
                    size_t base = (size_t)rowA * F + f0;
                    if (stat == 0) {
#pragma unroll
                        for (int i = 0; i < 16; ++i)
                            As[w][r][cbase + i] = (__bf16)(ssum[base + i] * idv * sc);
                    } else if (stat == 1) {
#pragma unroll
                        for (int i = 0; i < 16; ++i)
                            As[w][r][cbase + i] = (__bf16)((has ? fdec(smn[base + i]) : 0.f) * sc);
                    } else if (stat == 2) {
#pragma unroll
                        for (int i = 0; i < 16; ++i)
                            As[w][r][cbase + i] = (__bf16)((has ? fdec(smx[base + i]) : 0.f) * sc);
                    } else {
#pragma unroll
                        for (int i = 0; i < 16; ++i) {
                            float me = ssum[base + i] * idv, m2 = ssq[base + i] * idv;
                            float sd = sqrtf(fmaxf(m2 - me * me, 0.f) + 1e-5f);
                            As[w][r][cbase + i] = (__bf16)(sd * sc);
                        }
                    }
                }
            }
        } else {
            // generic per-element path (layer 0, F=4)
#pragma unroll
            for (int i = 0; i < 16; ++i) {
                int c = cbase + i;
                int k = k0 + c;
                float v = 0.0f;
                if (k < Kreal) {
                    if (MODE == MODE_EDGE) {
                        if (k < F)          v = A0[(size_t)dIdx * F + k];
                        else if (k < 2 * F) v = A0[(size_t)sIdx * F + (k - F)];
                        else {
                            int j = k - 2 * F;
                            v = eB[j] + ea0 * eW[j] + ea1 * eW[F + j]
                                      + ea2 * eW[2 * F + j] + ea3 * eW[3 * F + j];
                        }
                    } else { // MODE_FEAT
                        if (k < F) v = A0[(size_t)rowA * F + k];
                        else {
                            int kp   = k - F;
                            int grp  = kp / (4 * F);
                            int rem  = kp - grp * (4 * F);
                            int stat = rem / F;
                            int f    = rem - stat * F;
                            size_t idx = (size_t)rowA * F + f;
                            float val;
                            if (stat == 0)      val = ssum[idx] * idv;
                            else if (stat == 1) val = has ? fdec(smn[idx]) : 0.f;
                            else if (stat == 2) val = has ? fdec(smx[idx]) : 0.f;
                            else {
                                float me = ssum[idx] * idv, m2 = ssq[idx] * idv;
                                val = sqrtf(fmaxf(m2 - me * me, 0.f) + 1e-5f);
                            }
                            float sc = (grp == 0) ? 1.f : (grp == 1) ? scl1
                                     : (grp == 2) ? scl2 : scl3;
                            v = val * sc;
                        }
                    }
                }
                As[w][r][c] = (__bf16)v;
            }
        }

#if HAS_ASYNC_LDS
        async_wait0();
#endif
        __syncthreads();

        // ------------- load A fragment once, run NT WMMAs -------------
        v16bf a;
#pragma unroll
        for (int i = 0; i < 8; ++i) {
            a[i]     = As[w][M][kb + i];
            a[i + 8] = As[w][M][16 + kb + i];
        }
#pragma unroll
        for (int nt = 0; nt < NT; ++nt) {
            v16bf b;
#pragma unroll
            for (int i = 0; i < 8; ++i) {
                b[i]     = Bs[(nt << 4) + M][kb + i];
                b[i + 8] = Bs[(nt << 4) + M][16 + kb + i];
            }
            acc[nt] = __builtin_amdgcn_wmma_f32_16x16x32_bf16(false, a, false, b,
                                                              (short)0, acc[nt], false, false);
        }
        __syncthreads();
    }

    // ---- epilogue: C/D layout -> col = lane&15, rows M = (lane>>4)*8 + j ----
    int drow[8];
    if (MODE == MODE_EDGE) {
#pragma unroll
        for (int j = 0; j < 8; ++j) drow[j] = dst[row0 + ((lane >> 4) << 3) + j];
    }
#pragma unroll
    for (int nt = 0; nt < NT; ++nt) {
        int colg = (nt << 4) + (lane & 15);
        if (colg < Nout) {
            float bv = bias[colg];
#pragma unroll
            for (int j = 0; j < 8; ++j) {
                float v = acc[nt][j] + bv;
                if (MODE == MODE_EDGE) {
                    size_t idx = (size_t)drow[j] * F + colg;
                    atomicAdd(&osum[idx], v);
                    atomicAdd(&osq[idx], v * v);
                    atomicMin(&omn[idx], fenc(v));
                    atomicMax(&omx[idx], fenc(v));
                } else {
                    int row = row0 + ((lane >> 4) << 3) + j;
                    out[(size_t)row * HID + colg] = v;
                }
            }
        }
    }
}

// ------------------------------ BatchNorm ------------------------------
__global__ void k_bnstats(const float* __restrict__ y, float* __restrict__ mu,
                          float* __restrict__ rstd, int N) {
    __shared__ float s0[256], s1[256];
    int f = blockIdx.x;
    float s = 0.f, q = 0.f;
    for (int i = threadIdx.x; i < N; i += blockDim.x) {
        float v = y[(size_t)i * HID + f];
        s += v; q += v * v;
    }
    s0[threadIdx.x] = s; s1[threadIdx.x] = q;
    __syncthreads();
    for (int o = 128; o > 0; o >>= 1) {
        if ((int)threadIdx.x < o) { s0[threadIdx.x] += s0[threadIdx.x + o]; s1[threadIdx.x] += s1[threadIdx.x + o]; }
        __syncthreads();
    }
    if (threadIdx.x == 0) {
        float m = s0[0] / (float)N;
        float v = s1[0] / (float)N - m * m;
        mu[f] = m;
        rstd[f] = rsqrtf(v + 1e-5f);
    }
}

__global__ void k_bnapply(const float* __restrict__ y, const float* __restrict__ mu,
                          const float* __restrict__ rstd, const float* __restrict__ gamma,
                          const float* __restrict__ beta, float* __restrict__ hout, int N) {
    int i = blockIdx.x * blockDim.x + threadIdx.x;
    if (i >= N * HID) return;
    int f = i & (HID - 1);
    float v = gamma[f] * (y[i] - mu[f]) * rstd[f] + beta[f];
    hout[i] = leaky(v);
}

// ------------------------------ pooling + head ------------------------------
__global__ void k_gcount(const int* __restrict__ batch, float* __restrict__ gcount, int N) {
    int i = blockIdx.x * blockDim.x + threadIdx.x;
    if (i < N) atomicAdd(&gcount[batch[i]], 1.0f);
}
__global__ void k_pool(const float* __restrict__ h, const int* __restrict__ batch,
                       float* __restrict__ gsum, unsigned* __restrict__ gmx, int N) {
    int i = blockIdx.x * blockDim.x + threadIdx.x;
    if (i >= N * HID) return;
    int node = i >> 7, f = i & (HID - 1);
    int g = batch[node];
    float v = h[i];
    atomicAdd(&gsum[g * HID + f], v);
    atomicMax(&gmx[g * HID + f], fenc(v));
}
__global__ void k_gfin(const float* __restrict__ gsum, const unsigned* __restrict__ gmx,
                       const float* __restrict__ gcount, float* __restrict__ gfeat) {
    int i = blockIdx.x * blockDim.x + threadIdx.x;
    if (i >= N_GRAPHS * HID) return;
    int g = i >> 7, f = i & (HID - 1);
    float c = gcount[g];
    gfeat[g * 2 * HID + f]       = gsum[i] / fmaxf(c, 1.0f);
    gfeat[g * 2 * HID + HID + f] = (c > 0.f) ? fdec(gmx[i]) : 0.0f;
}
__global__ void k_head(const float* __restrict__ in, const float* __restrict__ W,
                       const float* __restrict__ b, float* __restrict__ out,
                       int Mrows, int K, int Nout, int act) {
    int i = blockIdx.x * blockDim.x + threadIdx.x;
    if (i >= Mrows * Nout) return;
    int m = i / Nout, n = i - m * Nout;
    float acc = b[n];
    for (int k = 0; k < K; ++k) acc += in[(size_t)m * K + k] * W[(size_t)k * Nout + n];
    out[i] = act ? leaky(acc) : acc;
}

// ------------------------------ launch ------------------------------
extern "C" void kernel_launch(void* const* d_in, const int* in_sizes, int n_in,
                              void* d_out, int out_size, void* d_ws, size_t ws_size,
                              hipStream_t stream) {
    (void)in_sizes; (void)n_in; (void)out_size; (void)ws_size;
    const float* x     = (const float*)d_in[0];
    const int*   ei    = (const int*)d_in[1];
    const float* eattr = (const float*)d_in[2];
    const int*   batch = (const int*)d_in[3];
    const int* src = ei;
    const int* dst = ei + N_EDGES;

    char* p = (char*)d_ws;
    auto alloc = [&](size_t bytes) -> char* {
        char* r = p; p += (bytes + 255) & ~(size_t)255; return r;
    };
    float*    hA    = (float*)alloc((size_t)N_NODES * HID * 4);
    float*    hB    = (float*)alloc((size_t)N_NODES * HID * 4);
    float*    y1    = (float*)alloc((size_t)N_NODES * HID * 4);
    float*    y2    = (float*)alloc((size_t)N_NODES * HID * 4);
    float*    ssum  = (float*)alloc((size_t)N_NODES * HID * 4);
    float*    ssq   = (float*)alloc((size_t)N_NODES * HID * 4);
    unsigned* smn   = (unsigned*)alloc((size_t)N_NODES * HID * 4);
    unsigned* smx   = (unsigned*)alloc((size_t)N_NODES * HID * 4);
    float*    cnt   = (float*)alloc((size_t)N_NODES * 4);
    float*    amp   = (float*)alloc((size_t)N_NODES * 4);
    float*    att   = (float*)alloc((size_t)N_NODES * 4);
    float*    lsc   = (float*)alloc((size_t)N_NODES * 4);
    float*    invd  = (float*)alloc((size_t)N_NODES * 4);
    float*    red   = (float*)alloc(256);
    float*    mu    = (float*)alloc(HID * 4);
    float*    rstd  = (float*)alloc(HID * 4);
    __bf16*   BtPre = (__bf16*)alloc((size_t)128 * 384 * 2);
    __bf16*   BtPost= (__bf16*)alloc((size_t)128 * 2176 * 2);
    __bf16*   BtLin = (__bf16*)alloc((size_t)128 * 128 * 2);
    float*    gsum  = (float*)alloc((size_t)N_GRAPHS * HID * 4);
    unsigned* gmx   = (unsigned*)alloc((size_t)N_GRAPHS * HID * 4);
    float*    gcnt  = (float*)alloc(N_GRAPHS * 4);
    float*    gfeat = (float*)alloc((size_t)N_GRAPHS * 2 * HID * 4);
    float*    t1    = (float*)alloc((size_t)N_GRAPHS * HID * 4);
    float*    t2    = (float*)alloc((size_t)N_GRAPHS * 64 * 4);

    auto cdiv = [](int a, int b) { return (a + b - 1) / b; };

    // ---- degree + PNA scalers ----
    k_fill_f<<<cdiv(N_NODES, 256), 256, 0, stream>>>(cnt, 0.f, N_NODES);
    k_degree<<<cdiv(N_EDGES, 256), 256, 0, stream>>>(dst, cnt, N_EDGES);
    k_fill_f<<<1, 256, 0, stream>>>(red, 0.f, 2);
    k_redstats<<<64, 256, 0, stream>>>(cnt, red, N_NODES);
    k_scalers<<<cdiv(N_NODES, 256), 256, 0, stream>>>(cnt, red, amp, att, lsc, invd, N_NODES);

    // ---- layers ----
    for (int l = 0; l < 6; ++l) {
        int F = (l == 0) ? 4 : HID;
        const float* hin  = (l == 0) ? x : (((l - 1) % 2 == 0) ? hA : hB);
        float*       hout = (l % 2 == 0) ? hA : hB;
        int base = 4 + 10 * l;
        const float* eW    = (const float*)d_in[base + 0];
        const float* eb    = (const float*)d_in[base + 1];
        const float* preW  = (const float*)d_in[base + 2];
        const float* preb  = (const float*)d_in[base + 3];
        const float* postW = (const float*)d_in[base + 4];
        const float* postb = (const float*)d_in[base + 5];
        const float* linW  = (const float*)d_in[base + 6];
        const float* linb  = (const float*)d_in[base + 7];
        const float* gamma = (const float*)d_in[base + 8];
        const float* beta  = (const float*)d_in[base + 9];

        int Kpre   = 3 * F;
        int KpreP  = ((Kpre + 31) / 32) * 32;
        int NpreP  = ((F + 15) / 16) * 16;
        int Kpost  = 17 * F;
        int KpostP = ((Kpost + 31) / 32) * 32;

        k_convW<<<cdiv(NpreP * KpreP, 256), 256, 0, stream>>>(preW, BtPre, Kpre, F, KpreP, NpreP);
        k_convW<<<cdiv(128 * KpostP, 256), 256, 0, stream>>>(postW, BtPost, Kpost, HID, KpostP, 128);
        k_convW<<<cdiv(128 * 128, 256), 256, 0, stream>>>(linW, BtLin, HID, HID, HID, 128);

        int NF = N_NODES * F;
        k_fill_f<<<cdiv(NF, 256), 256, 0, stream>>>(ssum, 0.f, NF);
        k_fill_f<<<cdiv(NF, 256), 256, 0, stream>>>(ssq, 0.f, NF);
        k_fill_u<<<cdiv(NF, 256), 256, 0, stream>>>(smn, 0xFFFFFFFFu, NF);
        k_fill_u<<<cdiv(NF, 256), 256, 0, stream>>>(smx, 0u, NF);

        int edgeBlocks = N_EDGES / 16 / 4;   // exact: 8192
        int nodeBlocks = N_NODES / 16 / 4;   // exact: 512
        if (l == 0) {
            // layer 0: Nout = 4 -> single 16-wide column tile
            k_gemm<MODE_EDGE, 1><<<edgeBlocks, 128, 0, stream>>>(
                hin, dst, src, eattr, eW, eb,
                nullptr, nullptr, nullptr, nullptr, nullptr, nullptr, nullptr, nullptr, nullptr,
                BtPre, preb, nullptr, ssum, ssq, smn, smx,
                F, Kpre, KpreP, F);
        } else {
            k_gemm<MODE_EDGE, 8><<<edgeBlocks, 128, 0, stream>>>(
                hin, dst, src, eattr, eW, eb,
                nullptr, nullptr, nullptr, nullptr, nullptr, nullptr, nullptr, nullptr, nullptr,
                BtPre, preb, nullptr, ssum, ssq, smn, smx,
                F, Kpre, KpreP, F);
        }
        k_gemm<MODE_FEAT, 8><<<nodeBlocks, 128, 0, stream>>>(
            hin, nullptr, nullptr, nullptr, nullptr, nullptr,
            ssum, ssq, smn, smx, amp, att, lsc, invd, cnt,
            BtPost, postb, y1, nullptr, nullptr, nullptr, nullptr,
            F, Kpost, KpostP, HID);
        k_gemm<MODE_PLAIN, 8><<<nodeBlocks, 128, 0, stream>>>(
            y1, nullptr, nullptr, nullptr, nullptr, nullptr,
            nullptr, nullptr, nullptr, nullptr, nullptr, nullptr, nullptr, nullptr, nullptr,
            BtLin, linb, y2, nullptr, nullptr, nullptr, nullptr,
            F, HID, HID, HID);

        k_bnstats<<<HID, 256, 0, stream>>>(y2, mu, rstd, N_NODES);
        k_bnapply<<<cdiv(N_NODES * HID, 256), 256, 0, stream>>>(y2, mu, rstd, gamma, beta, hout, N_NODES);
    }

    // ---- graph pooling ----
    const float* hfin = hB;  // layer 5 output
    k_fill_f<<<cdiv(N_GRAPHS * HID, 256), 256, 0, stream>>>(gsum, 0.f, N_GRAPHS * HID);
    k_fill_u<<<cdiv(N_GRAPHS * HID, 256), 256, 0, stream>>>(gmx, 0u, N_GRAPHS * HID);
    k_fill_f<<<1, 256, 0, stream>>>(gcnt, 0.f, N_GRAPHS);
    k_gcount<<<cdiv(N_NODES, 256), 256, 0, stream>>>(batch, gcnt, N_NODES);
    k_pool<<<cdiv(N_NODES * HID, 256), 256, 0, stream>>>(hfin, batch, gsum, gmx, N_NODES);
    k_gfin<<<cdiv(N_GRAPHS * HID, 256), 256, 0, stream>>>(gsum, gmx, gcnt, gfeat);

    // ---- MLP head (tiny: 64 graphs) ----
    const float* W1 = (const float*)d_in[64];
    const float* b1 = (const float*)d_in[65];
    const float* W2 = (const float*)d_in[66];
    const float* b2 = (const float*)d_in[67];
    const float* W3 = (const float*)d_in[68];
    const float* b3 = (const float*)d_in[69];
    k_head<<<cdiv(N_GRAPHS * HID, 256), 256, 0, stream>>>(gfeat, W1, b1, t1, N_GRAPHS, 2 * HID, HID, 1);
    k_head<<<cdiv(N_GRAPHS * 64, 256), 256, 0, stream>>>(t1, W2, b2, t2, N_GRAPHS, HID, 64, 1);
    k_head<<<1, 64, 0, stream>>>(t2, W3, b3, (float*)d_out, N_GRAPHS, 64, 1, 0);
}
